// GRU_58480274702939
// MI455X (gfx1250) — matrix operand order repaired and checked by reading, not compiled, not run on hardware
//
#include <hip/hip_runtime.h>
#include <hip/hip_bf16.h>

#define SEQ   512
#define BATCH 64
#define DIM   1024
#define BD    (BATCH * DIM)

#define NBLOCKS  64
#define NTHREADS 128           // 4 wave32 per block
#define WPAD     1032          // padded LDS row stride (bf16 elems) to spread banks
#define SMEM_BYTES (3 * 16 * WPAD * 2)   // 3 weight slabs of 16 cols x 1024 k

typedef __attribute__((ext_vector_type(16))) __bf16 bf16x16;
typedef __attribute__((ext_vector_type(8)))  float  floatx8;

union Frag {
  bf16x16 v;
  uint4   u[2];
};

// ---- workspace layout (bytes) ----
#define WS_CTRL  0                  // 2 x u32 barrier state
#define WS_H     256                // BD f32  (h state)
#define WS_Z     (WS_H + BD * 4)    // BD f32  (z gate)
#define WS_U     (WS_Z + BD * 4)    // BD bf16 (x_t + h)
#define WS_V     (WS_U + BD * 2)    // BD bf16 (x_t + r*h)

// Fast activations: single v_rcp_f32 instead of IEEE divide sequence.
__device__ __forceinline__ float sigmoidf_dev(float x) {
  return __builtin_amdgcn_rcpf(1.0f + __expf(-x));
}
__device__ __forceinline__ float tanhf_dev(float y) {
  // tanh(y) = 1 - 2/(exp(2y)+1); saturates correctly at +/-inf.
  return 1.0f - 2.0f * __builtin_amdgcn_rcpf(__expf(2.0f * y) + 1.0f);
}

// Grid-wide sense barrier (agent scope).
__device__ __forceinline__ void grid_sync(unsigned* cnt, unsigned* gen) {
  __syncthreads();
  if (threadIdx.x == 0) {
    __threadfence();
    unsigned g = __hip_atomic_load(gen, __ATOMIC_RELAXED, __HIP_MEMORY_SCOPE_AGENT);
    unsigned a = __hip_atomic_fetch_add(cnt, 1u, __ATOMIC_ACQ_REL, __HIP_MEMORY_SCOPE_AGENT);
    if (a == (unsigned)(NBLOCKS - 1)) {
      __hip_atomic_store(cnt, 0u, __ATOMIC_RELAXED, __HIP_MEMORY_SCOPE_AGENT);
      __hip_atomic_fetch_add(gen, 1u, __ATOMIC_RELEASE, __HIP_MEMORY_SCOPE_AGENT);
    } else {
      while (__hip_atomic_load(gen, __ATOMIC_ACQUIRE, __HIP_MEMORY_SCOPE_AGENT) == g) {
        __builtin_amdgcn_s_sleep(2);
      }
    }
    __threadfence();
  }
  __syncthreads();
}

// Prep: h = 0, u = bf16(x[0]), reset barrier words.
__global__ void gru_prep(const float* __restrict__ x, unsigned char* __restrict__ ws) {
  unsigned* ctrl = reinterpret_cast<unsigned*>(ws + WS_CTRL);
  float*  h = reinterpret_cast<float*>(ws + WS_H);
  __bf16* u = reinterpret_cast<__bf16*>(ws + WS_U);
  const size_t tid = (size_t)blockIdx.x * blockDim.x + threadIdx.x;
  const size_t nth = (size_t)gridDim.x * blockDim.x;
  if (tid == 0) { ctrl[0] = 0u; ctrl[1] = 0u; }
  for (size_t i = tid; i < (size_t)BD; i += nth) {
    h[i] = 0.0f;
    u[i] = (__bf16)x[i];     // u_0 = x_0 + h_0, h_0 = 0
  }
}

// Persistent recurrent kernel: weights LDS-resident, 512 steps, 2 grid syncs/step.
// Block b owns output columns [16b, 16b+16); wave w owns m-tile w (rows 16w..16w+15).
__global__ void __launch_bounds__(NTHREADS, 1)
gru_persistent(const float* __restrict__ x,
               const float* __restrict__ Wz,
               const float* __restrict__ Wr,
               const float* __restrict__ W,
               const float* __restrict__ bias,
               float* __restrict__ out,
               unsigned char* __restrict__ ws) {
  extern __shared__ unsigned char smem[];
  __bf16* wz_l = reinterpret_cast<__bf16*>(smem);
  __bf16* wr_l = wz_l + 16 * WPAD;
  __bf16* wc_l = wr_l + 16 * WPAD;

  unsigned* cnt = reinterpret_cast<unsigned*>(ws + WS_CTRL);
  unsigned* gen = cnt + 1;
  float*  h = reinterpret_cast<float*>(ws + WS_H);
  float*  z = reinterpret_cast<float*>(ws + WS_Z);
  __bf16* u = reinterpret_cast<__bf16*>(ws + WS_U);
  __bf16* v = reinterpret_cast<__bf16*>(ws + WS_V);

  const int lane  = threadIdx.x & 31;
  const int mtile = threadIdx.x >> 5;            // 0..3  (rows 16*mtile..)
  const int cb    = blockIdx.x * 16;             // column slab base

  // ---- prologue: convert this block's 16-column slab of all 3 weights into LDS ----
  for (int idx = threadIdx.x; idx < 16 * 1024; idx += NTHREADS) {
    const int r = idx >> 10;          // local column (output latent index cb+r)
    const int c = idx & 1023;         // k index
    const size_t g = (size_t)(cb + r) * DIM + c;
    const int l = r * WPAD + c;
    wz_l[l] = (__bf16)Wz[g];
    wr_l[l] = (__bf16)Wr[g];
    wc_l[l] = (__bf16)W[g];
  }
  __syncthreads();

  // ---- hoisted per-thread geometry ----
  const int mbase = mtile * 16;
  const int colL  = lane & 15;                   // local column
  const int col   = cb + colL;                   // global output column
  const int ak    = (lane >> 4) * 16;            // A-frag K sub-offset
  const int bk    = (lane >> 4) * 16;            // B-frag K sub-offset
  const int rb    = mbase + ((lane >> 4) << 3);  // first of 8 accumulator rows
  const size_t arow_off = (size_t)(mbase + colL) * DIM + ak;  // A row uses lane&15 as M
  const __bf16* bz = wz_l + colL * WPAD + bk;
  const __bf16* br = wr_l + colL * WPAD + bk;
  const __bf16* bc = wc_l + colL * WPAD + bk;
  const float bb = 2.0f * bias[col];

  for (int s = 0; s < SEQ; ++s) {
    const float* xt = x + (size_t)s * BD;
    // ---------- Phase A: z_pre = u@Wz^T, r_pre = u@Wr^T (A-frag shared) ----------
    {
      floatx8 accz = {}, accr = {};
      const __bf16* au = u + arow_off;
#pragma unroll 4
      for (int k = 0; k < DIM; k += 32) {
        Frag a, fz, fr;
        a.u[0]  = *reinterpret_cast<const uint4*>(au + k);
        a.u[1]  = *reinterpret_cast<const uint4*>(au + k + 8);
        fz.u[0] = *reinterpret_cast<const uint4*>(bz + k);
        fz.u[1] = *reinterpret_cast<const uint4*>(bz + k + 8);
        fr.u[0] = *reinterpret_cast<const uint4*>(br + k);
        fr.u[1] = *reinterpret_cast<const uint4*>(br + k + 8);
        accz = __builtin_amdgcn_wmma_f32_16x16x32_bf16(false, a.v, false, fz.v,
                                                       (short)0, accz, false, false);
        accr = __builtin_amdgcn_wmma_f32_16x16x32_bf16(false, a.v, false, fr.v,
                                                       (short)0, accr, false, false);
      }
#pragma unroll
      for (int j = 0; j < 8; ++j) {
        const int i0 = (rb + j) * DIM + col;
        z[i0] = sigmoidf_dev(accz[j]);
        v[i0] = (__bf16)(xt[i0] + sigmoidf_dev(accr[j]) * h[i0]);
      }
    }
    grid_sync(cnt, gen);

    // ---------- Phase B: h~ = tanh(v@W^T + 2b); h = (1-z)h + z h~ ----------
    {
      floatx8 accc = {};
      const __bf16* av = v + arow_off;
#pragma unroll 4
      for (int k = 0; k < DIM; k += 32) {
        Frag a, fc;
        a.u[0]  = *reinterpret_cast<const uint4*>(av + k);
        a.u[1]  = *reinterpret_cast<const uint4*>(av + k + 8);
        fc.u[0] = *reinterpret_cast<const uint4*>(bc + k);
        fc.u[1] = *reinterpret_cast<const uint4*>(bc + k + 8);
        accc = __builtin_amdgcn_wmma_f32_16x16x32_bf16(false, a.v, false, fc.v,
                                                       (short)0, accc, false, false);
      }
      const float* xn = x + (size_t)(s + 1) * BD;
#pragma unroll
      for (int j = 0; j < 8; ++j) {
        const int i0 = (rb + j) * DIM + col;
        const float ht = tanhf_dev(accc[j] + bb);
        const float zz = z[i0];
        const float hn = (1.0f - zz) * h[i0] + zz * ht;
        out[(size_t)s * BD + i0] = hn;
        h[i0] = hn;
        if (s + 1 < SEQ) {
          u[i0] = (__bf16)(xn[i0] + hn);   // u_{t+1} = x_{t+1} + h_{t+1}
        }
      }
    }
    grid_sync(cnt, gen);
  }
}

extern "C" void kernel_launch(void* const* d_in, const int* in_sizes, int n_in,
                              void* d_out, int out_size, void* d_ws, size_t ws_size,
                              hipStream_t stream) {
  const float* x  = (const float*)d_in[0];
  const float* Wz = (const float*)d_in[1];
  const float* Wr = (const float*)d_in[2];
  const float* W  = (const float*)d_in[3];
  const float* b  = (const float*)d_in[4];
  float* out = (float*)d_out;
  unsigned char* ws = (unsigned char*)d_ws;

  gru_prep<<<256, 256, 0, stream>>>(x, ws);
  gru_persistent<<<NBLOCKS, NTHREADS, SMEM_BYTES, stream>>>(x, Wz, Wr, W, b, out, ws);
}
